// CausalSelfAttention_48490180772071
// MI455X (gfx1250) — matrix operand order repaired and checked by reading, not compiled, not run on hardware
//
#include <hip/hip_runtime.h>
#include <hip/hip_bf16.h>

// ---------------------------------------------------------------------------
// CDNA5 (gfx1250) causal self-attention forward:
//   qkv = x @ W_attn + b_attn            (bf16 WMMA; q columns pre-scaled 1/8)
//   y   = flash_attention(q, k, v)       (bf16 WMMA QK^T, PV and P@1, fp32 max)
//   out = y @ W_proj + b_proj            (bf16 WMMA, fp32 out; A tile via TDM)
// B=4, S=2048, D=1024, H=16, HD=64
// ---------------------------------------------------------------------------

typedef __attribute__((ext_vector_type(16))) __bf16 v16bf;
typedef __attribute__((ext_vector_type(8)))  float  v8f;

union BF16x16 {
    unsigned int u[8];
    uint4        q[2];
    v16bf        v;
    unsigned short h[16];
};

static __device__ __forceinline__ unsigned short f2bf(float f) {
    unsigned int u = __float_as_uint(f);
    u += 0x7FFFu + ((u >> 16) & 1u);        // round-to-nearest-even
    return (unsigned short)(u >> 16);
}

static __device__ __forceinline__ v8f wmma_bf16(v16bf a, v16bf b, v8f c) {
    // D = A(16x32 bf16) * B(32x16 bf16) + C(16x16 f32)
    return __builtin_amdgcn_wmma_f32_16x16x32_bf16(
        /*neg_a=*/false, a, /*neg_b=*/false, b,
        /*c_mod=*/(short)0, c, /*reuse_a=*/false, /*reuse_b=*/false);
}

// Same, but hint the matrix pipe that the NEXT wmma reuses this A operand.
static __device__ __forceinline__ v8f wmma_bf16_ra(v16bf a, v16bf b, v8f c) {
    return __builtin_amdgcn_wmma_f32_16x16x32_bf16(
        false, a, false, b, (short)0, c, /*reuse_a=*/true, /*reuse_b=*/false);
}

#if defined(__gfx1250__)
typedef __attribute__((ext_vector_type(4))) unsigned int v4u;
typedef __attribute__((ext_vector_type(4))) int v4i;
typedef __attribute__((ext_vector_type(8))) int v8i;

// LDS byte offset of a __shared__ object (addrspace(3) pointer value).
static __device__ __forceinline__ unsigned lds_offset_of(const void* p) {
    return (unsigned)(unsigned long long)
        (__attribute__((address_space(3))) const void*)p;
}

// TDM: 2D tile of bf16 (2-byte) elements, global -> LDS, with optional LDS row
// padding.  pad_interval_code: pad after 2^(code+1) DWORDs; pad_amount_code:
// pad (code+1) DWORDs.
static __device__ __forceinline__ void tdm_load_2d_bf16(
    unsigned long long gaddr, unsigned lds_off,
    unsigned tensor_w, unsigned tensor_h, unsigned long long row_stride_elems,
    unsigned tile_w, unsigned tile_h,
    unsigned pad_interval_code, unsigned pad_amount_code, bool pad_en)
{
    v4u g0;
    g0.x = 1u;                                         // count=1, user D#
    g0.y = lds_off;                                    // lds_addr [63:32]
    g0.z = (unsigned)(gaddr & 0xFFFFFFFFull);          // global_addr lo
    g0.w = (unsigned)((gaddr >> 32) & 0x01FFFFFFull)   // global_addr hi
         | (2u << 30);                                 // type=2 ("image")

    v8i g1;
    unsigned w0 = (1u << 16);                          // data_size=1 -> 2B
    if (pad_en)
        w0 |= (1u << 20)                               // pad_enable
            | ((pad_interval_code & 7u) << 22)
            | ((pad_amount_code & 127u) << 25);
    g1[0] = (int)w0;
    g1[1] = (int)((tensor_w & 0xFFFFu) << 16);         // tensor_dim0[15:0]
    g1[2] = (int)(((tensor_w >> 16) & 0xFFFFu)         // tensor_dim0[31:16]
                | ((tensor_h & 0xFFFFu) << 16));       // tensor_dim1[15:0]
    g1[3] = (int)(((tensor_h >> 16) & 0xFFFFu)         // tensor_dim1[31:16]
                | ((tile_w & 0xFFFFu) << 16));         // tile_dim0
    g1[4] = (int)(tile_h & 0xFFFFu);                   // tile_dim1 (tile_dim2=0)
    g1[5] = (int)(unsigned)(row_stride_elems & 0xFFFFFFFFull);  // dim0_stride lo
    g1[6] = (int)(unsigned)((row_stride_elems >> 32) & 0xFFFFull);
    g1[7] = 0;

    v4i g2 = {0, 0, 0, 0};
    v4i g3 = {0, 0, 0, 0};
#if __clang_major__ >= 23
    v8i g4 = {0, 0, 0, 0, 0, 0, 0, 0};
    __builtin_amdgcn_tensor_load_to_lds(g0, g1, g2, g3, g4, 0);
#else
    __builtin_amdgcn_tensor_load_to_lds(g0, g1, g2, g3, 0);
#endif
}
#endif // __gfx1250__

// ---------------------------------------------------------------------------
// Tiled GEMM: Out[M,N] = A[M,K] @ (W[K,N] * colscale) + bias[N] * colscale
// colscale = wscale for columns < wscale_cols, else 1.0 (per-tile uniform).
// A is fp32 or bf16 (A_F32); Out is bf16 or fp32 (OUT_BF16). W/bias are fp32.
// Block 256 threads (8 waves), tile 128x128, k-step 32.
// bf16-A tiles are staged by the Tensor Data Mover; the DMA overlaps with the
// fp32->bf16 W-tile conversion before the TENSORcnt wait.
// ---------------------------------------------------------------------------
template <bool A_F32, bool OUT_BF16>
__global__ void __launch_bounds__(256)
gemm_bias_kernel(const void* __restrict__ Ap, const float* __restrict__ W,
                 const float* __restrict__ bias, void* __restrict__ Outp,
                 int M, int N, int K, float wscale, int wscale_cols)
{
    constexpr int BM = 128, BN = 128, BK = 32, LDT = BK + 8; // 40 halfs/row
    __shared__ __align__(16) unsigned short sA[BM * LDT];     // [m][k] bf16
    __shared__ __align__(16) unsigned short sB[BN * LDT];     // [n][k] bf16 (transposed)

    const int tid    = threadIdx.x;
    const int lane   = tid & 31;
    const int wid    = tid >> 5;
    const int l16    = lane & 15;
    const int half16 = lane >> 4;
    const int wm     = (wid >> 1) * 32;   // wave rows within tile
    const int wn     = (wid & 1) * 64;    // wave cols within tile
    const int row0   = blockIdx.y * BM;
    const int col0   = blockIdx.x * BN;

    // Per-tile uniform column scale (q/k/v boundary is tile-aligned).
    const float cs = (col0 < wscale_cols) ? wscale : 1.0f;

    v8f acc[2][4];
    const v8f vzero = {};
#pragma unroll
    for (int mi = 0; mi < 2; ++mi)
#pragma unroll
        for (int ni = 0; ni < 4; ++ni) acc[mi][ni] = vzero;

#if defined(__gfx1250__)
    const unsigned sA_off = lds_offset_of(&sA[0]);
#endif

    for (int k0 = 0; k0 < K; k0 += BK) {
        __syncthreads();
        // ---- stage A tile ----
        if constexpr (A_F32) {
            const float* A = (const float*)Ap;
#pragma unroll
            for (int i = 0; i < 4; ++i) {
                int e = (i * 256 + tid) * 4;           // 128*32 elems
                int r = e >> 5, c = e & 31;
                float4 f = *(const float4*)(A + (size_t)(row0 + r) * K + k0 + c);
                ushort4 hv;
                hv.x = f2bf(f.x); hv.y = f2bf(f.y);
                hv.z = f2bf(f.z); hv.w = f2bf(f.w);
                *(ushort4*)(&sA[r * LDT + c]) = hv;
            }
        } else {
#if defined(__gfx1250__)
            // Tensor Data Mover: 128x32 bf16 tile, LDS pitch 40 halfs via
            // padding: pad 4 DWORDs (code 3) after every 16 DWORDs (code 3).
            if (wid == 0) {
                const unsigned short* gsrc =
                    (const unsigned short*)Ap + (size_t)row0 * K + k0;
                tdm_load_2d_bf16((unsigned long long)(const void*)gsrc, sA_off,
                                 (unsigned)K, (unsigned)M,
                                 (unsigned long long)K,
                                 (unsigned)BK, (unsigned)BM,
                                 /*interval 16 dw*/ 3u, /*amount 4 dw*/ 3u, true);
            }
#else
            const unsigned short* A = (const unsigned short*)Ap;
#pragma unroll
            for (int i = 0; i < 2; ++i) {
                int e = (i * 256 + tid) * 8;
                int r = e >> 5, c = e & 31;
                uint4 qv = *(const uint4*)(A + (size_t)(row0 + r) * K + k0 + c);
                *(uint4*)(&sA[r * LDT + c]) = qv;
            }
#endif
        }
        // ---- stage W tile transposed: sB[n][k] = bf16(cs * W[k0+k][col0+n]) ----
        // (TDM for the A tile runs underneath this conversion work.)
#pragma unroll
        for (int i = 0; i < 4; ++i) {
            int e  = (i * 256 + tid) * 4;              // 32*128 elems
            int kr = e >> 7, nc = e & 127;
            float4 f = *(const float4*)(W + (size_t)(k0 + kr) * N + col0 + nc);
            sB[(nc + 0) * LDT + kr] = f2bf(f.x * cs);
            sB[(nc + 1) * LDT + kr] = f2bf(f.y * cs);
            sB[(nc + 2) * LDT + kr] = f2bf(f.z * cs);
            sB[(nc + 3) * LDT + kr] = f2bf(f.w * cs);
        }
#if defined(__gfx1250__)
        if constexpr (!A_F32) {
            if (wid == 0) __builtin_amdgcn_s_wait_tensorcnt(0);
        }
#endif
        __syncthreads();

        // ---- fragments in wave32 WMMA layout ----
        BF16x16 afr[2], bfr[4];
#pragma unroll
        for (int mi = 0; mi < 2; ++mi) {
            // A 16x32: lane holds row m=l16, K = half16*8+[0..7], 16+half16*8+[0..7]
            const unsigned short* p = &sA[(wm + mi * 16 + l16) * LDT + half16 * 8];
            afr[mi].q[0] = *(const uint4*)(p);
            afr[mi].q[1] = *(const uint4*)(p + 16);
        }
#pragma unroll
        for (int ni = 0; ni < 4; ++ni) {
            // B 32x16: lane holds col n=l16, K = half16*8+[0..7], 16+half16*8+[0..7]
            const unsigned short* p = &sB[(wn + ni * 16 + l16) * LDT + half16 * 8];
            bfr[ni].q[0] = *(const uint4*)(p);
            bfr[ni].q[1] = *(const uint4*)(p + 16);
        }
#pragma unroll
        for (int mi = 0; mi < 2; ++mi)
#pragma unroll
            for (int ni = 0; ni < 4; ++ni)
                acc[mi][ni] = wmma_bf16(afr[mi].v, bfr[ni].v, acc[mi][ni]);
    }

    // ---- epilogue: C layout m = r + 8*half16, n = l16 ----
#pragma unroll
    for (int mi = 0; mi < 2; ++mi) {
#pragma unroll
        for (int ni = 0; ni < 4; ++ni) {
            int col = col0 + wn + ni * 16 + l16;
            float bv = bias[col] * cs;
#pragma unroll
            for (int r = 0; r < 8; ++r) {
                int row = row0 + wm + mi * 16 + r + half16 * 8;
                float v = acc[mi][ni][r] + bv;
                if constexpr (OUT_BF16)
                    ((unsigned short*)Outp)[(size_t)row * N + col] = f2bf(v);
                else
                    ((float*)Outp)[(size_t)row * N + col] = v;
            }
        }
    }
}

// ---------------------------------------------------------------------------
// Flash attention (causal). qkv: bf16 [B*S][3*D] with q pre-scaled by 1/8;
// y: bf16 [B*S][D].
// Block = 128 threads (4 waves); each block owns 64 q rows of one (b,h);
// each wave owns 16 q rows. K-loop in 32-key tiles up to causal bound.
// Softmax denominator computed as P @ ones via WMMA (no shuffle-sum).
// ---------------------------------------------------------------------------
__global__ void __launch_bounds__(128)
flash_attn_kernel(const unsigned short* __restrict__ qkv,
                  unsigned short* __restrict__ y)
{
    constexpr int S = 2048, D = 1024, H = 16, HD = 64;
    constexpr int QLD = 3 * D;          // 3072
    constexpr int LDV = 40;             // padded LDS row stride (halfs)

    __shared__ __align__(16) unsigned short sVt[HD * LDV];      // V^T [hd][key]
    __shared__ __align__(16) unsigned short sP[4][16 * LDV];    // per-wave P

    const int tid    = threadIdx.x;
    const int lane   = tid & 31;
    const int wid    = tid >> 5;
    const int l16    = lane & 15;
    const int half16 = lane >> 4;

    const int bh = blockIdx.y;
    const int b  = bh / H, h = bh % H;
    const int q0 = blockIdx.x * 64;
    const int qw = q0 + wid * 16;

    const size_t rowbase = (size_t)b * S;
    const unsigned short* Qg = qkv + rowbase * QLD + h * HD;
    const unsigned short* Kg = qkv + rowbase * QLD + D + h * HD;
    const unsigned short* Vg = qkv + rowbase * QLD + 2 * D + h * HD;

    // Q fragments: 16x64 as two 16x32 A-fragments, straight from global.
    BF16x16 qf[2];
#pragma unroll
    for (int f = 0; f < 2; ++f) {
        const unsigned short* p = Qg + (size_t)(qw + l16) * QLD + f * 32 + half16 * 8;
        qf[f].q[0] = *(const uint4*)(p);
        qf[f].q[1] = *(const uint4*)(p + 16);
    }

    // All-ones B fragment: ol = P @ 1 gives the softmax row sums.
    BF16x16 ones;
#pragma unroll
    for (int i = 0; i < 8; ++i) ones.u[i] = 0x3F803F80u;   // bf16 1.0 pair

    const v8f vzero = {};
    v8f o[4];
#pragma unroll
    for (int n = 0; n < 4; ++n) o[n] = vzero;
    v8f ol = vzero;                      // running row-sum accumulator
    float mrun[8];
#pragma unroll
    for (int r = 0; r < 8; ++r) mrun[r] = -3.0e38f;

    const int ksteps = (q0 + 64) / 32;  // causal bound (uniform per block)

    for (int kt = 0; kt < ksteps; ++kt) {
        const int ks0 = kt * 32;
        __syncthreads();
        // ---- stage V (32 keys x 64 hd) transposed into LDS ----
#pragma unroll
        for (int i = 0; i < 2; ++i) {
            int e  = (i * 128 + tid) * 8;          // 32*64 elems
            int kr = e >> 6, nc = e & 63;
            uint4 v4 = *(const uint4*)(Vg + (size_t)(ks0 + kr) * QLD + nc);
            const unsigned short* hs = (const unsigned short*)&v4;
#pragma unroll
            for (int j = 0; j < 8; ++j)
                sVt[(nc + j) * LDV + kr] = hs[j];
        }
        // prefetch next V tile while this one is consumed
        if (kt + 1 < ksteps)
            __builtin_prefetch(Vg + (size_t)(ks0 + 32 + (tid >> 2)) * QLD, 0, 1);
        __syncthreads();

        if (ks0 > qw + 15) continue;   // tile fully masked for this wave's rows

        // ---- S = Q @ K^T : batch all K^T fragment loads, then 4 WMMAs ----
        BF16x16 kf[2][2];
#pragma unroll
        for (int j = 0; j < 2; ++j) {
            const unsigned short* pb = Kg + (size_t)(ks0 + j * 16 + l16) * QLD;
#pragma unroll
            for (int f = 0; f < 2; ++f) {
                const unsigned short* p = pb + f * 32 + half16 * 8;
                kf[j][f].q[0] = *(const uint4*)(p);
                kf[j][f].q[1] = *(const uint4*)(p + 16);
            }
        }
        v8f sc[2];
#pragma unroll
        for (int j = 0; j < 2; ++j) {
            sc[j] = vzero;
#pragma unroll
            for (int f = 0; f < 2; ++f)
                sc[j] = wmma_bf16(qf[f].v, kf[j][f].v, sc[j]);
        }

        // ---- causal mask only on diagonal-straddling tiles (q pre-scaled) ----
        const bool needMask = (ks0 + 31) > qw;
        if (needMask) {
#pragma unroll
            for (int j = 0; j < 2; ++j)
#pragma unroll
                for (int r = 0; r < 8; ++r) {
                    int qrow = qw + r + half16 * 8;
                    int key  = ks0 + j * 16 + l16;
                    sc[j][r] = (key <= qrow) ? sc[j][r] : -3.0e38f;
                }
        }

        // ---- online softmax: row max via lane shuffles; sums via P@1 ----
#pragma unroll
        for (int r = 0; r < 8; ++r) {
            float m = fmaxf(sc[0][r], sc[1][r]);
#pragma unroll
            for (int off = 8; off >= 1; off >>= 1)
                m = fmaxf(m, __shfl_xor(m, off, 32));
            float mnew = fmaxf(mrun[r], m);
            sc[0][r] = __expf(sc[0][r] - mnew);
            sc[1][r] = __expf(sc[1][r] - mnew);
            float corr = __expf(mrun[r] - mnew);
            mrun[r] = mnew;
            ol[r] *= corr;
#pragma unroll
            for (int n = 0; n < 4; ++n) o[n][r] *= corr;
        }

        // ---- P: C-layout -> bf16 -> LDS -> A-layout fragment ----
        unsigned short* Pw = &sP[wid][0];
#pragma unroll
        for (int j = 0; j < 2; ++j)
#pragma unroll
            for (int r = 0; r < 8; ++r)
                Pw[(r + half16 * 8) * LDV + j * 16 + l16] = f2bf(sc[j][r]);

        BF16x16 pf;
        {
            const unsigned short* p = Pw + l16 * LDV + half16 * 8;
            pf.q[0] = *(const uint4*)(p);
            pf.q[1] = *(const uint4*)(p + 16);
        }

        // ---- ol += P @ 1 and O += P @ V; shared-A chain uses reuse_a hint ----
        ol = wmma_bf16_ra(pf.v, ones.v, ol);
#pragma unroll
        for (int n = 0; n < 4; ++n) {
            BF16x16 vf;
            const unsigned short* p = &sVt[(n * 16 + l16) * LDV + half16 * 8];
            vf.q[0] = *(const uint4*)(p);
            vf.q[1] = *(const uint4*)(p + 16);
            o[n] = (n < 3) ? wmma_bf16_ra(pf.v, vf.v, o[n])
                           : wmma_bf16(pf.v, vf.v, o[n]);
        }
    }

    // ---- epilogue: y = O / rowsum ----
#pragma unroll
    for (int r = 0; r < 8; ++r) {
        float inv = 1.0f / ol[r];
        int row = qw + r + half16 * 8;
#pragma unroll
        for (int n = 0; n < 4; ++n) {
            int col = h * HD + n * 16 + l16;
            y[(rowbase + row) * D + col] = f2bf(o[n][r] * inv);
        }
    }
}

// ---------------------------------------------------------------------------
extern "C" void kernel_launch(void* const* d_in, const int* in_sizes, int n_in,
                              void* d_out, int out_size, void* d_ws, size_t ws_size,
                              hipStream_t stream)
{
    (void)in_sizes; (void)n_in; (void)out_size; (void)ws_size;
    const float* x      = (const float*)d_in[0];
    const float* W_attn = (const float*)d_in[1];
    const float* b_attn = (const float*)d_in[2];
    const float* W_proj = (const float*)d_in[3];
    const float* b_proj = (const float*)d_in[4];
    float* out = (float*)d_out;

    constexpr int Bc = 4, Sc = 2048, Dc = 1024, Hc = 16;
    constexpr int M = Bc * Sc;                       // 8192 rows

    unsigned short* qkv = (unsigned short*)d_ws;                         // [M][3D] bf16 (48 MB)
    unsigned short* yb  = (unsigned short*)d_ws + (size_t)M * 3 * Dc;    // [M][D]  bf16 (16 MB)

    // 1) qkv = x @ W_attn + b_attn; q columns (0..D) pre-scaled by 1/sqrt(HD)
    {
        dim3 grid((3 * Dc) / 128, M / 128);
        gemm_bias_kernel<true, true><<<grid, 256, 0, stream>>>(
            x, W_attn, b_attn, qkv, M, 3 * Dc, Dc, 0.125f, Dc);
    }
    // 2) flash attention (bf16 in/out)
    {
        dim3 grid(Sc / 64, Bc * Hc);
        flash_attn_kernel<<<grid, 128, 0, stream>>>(qkv, yb);
    }
    // 3) out = y @ W_proj + b_proj  (bf16 in, fp32 out; A tile via TDM)
    {
        dim3 grid(Dc / 128, M / 128);
        gemm_bias_kernel<false, false><<<grid, 256, 0, stream>>>(
            yb, W_proj, b_proj, out, M, Dc, Dc, 1.0f, 0);
    }
}
